// SelfAttention_76622216561163
// MI455X (gfx1250) — compile-verified, matrix operand order checked by reading, not run on hardware
//
#include <hip/hip_runtime.h>

// ---------------------------------------------------------------------------
// Compressive-Transformer attention for MI455X (gfx1250, wave32, WMMA).
// All matrix math on v_wmma_f32_16x16x32_bf16 (bf16 in, f32 accumulate).
// GEMM B-tiles staged with async global->LDS copies (ASYNCcnt) + double
// buffering so the fill of tile k+1 overlaps the WMMAs of tile k.
// ---------------------------------------------------------------------------

typedef __attribute__((ext_vector_type(16))) __bf16 v16bf;
typedef __attribute__((ext_vector_type(8)))  float  v8f;

union Frag { v16bf v; uint4 q[2]; };

__device__ __forceinline__ unsigned short f2bf(float f) {
  unsigned int u = __float_as_uint(f);
  unsigned int r = u + 0x7FFFu + ((u >> 16) & 1u);   // round-to-nearest-even
  return (unsigned short)(r >> 16);
}

__device__ __forceinline__ v8f wmma_bf16(v16bf a, v16bf b, v8f c) {
  return __builtin_amdgcn_wmma_f32_16x16x32_bf16(false, a, false, b,
                                                 (short)0, c, false, false);
}

// Loads one 16x32 operand fragment half for this lane: elements [0..7] and
// [16..23] relative to p (p already includes row, k0 and half*8 offsets).
__device__ __forceinline__ Frag load_frag_g(const unsigned short* p) {
  Frag f;
  f.q[0] = *(const uint4*)(p);
  f.q[1] = *(const uint4*)(p + 16);
  return f;
}

// Async copy of 32 bytes (2 x b128, 32B apart) global -> LDS.  The immediate
// offset is applied to BOTH the LDS destination (VDST + IOFFSET) and the
// global source (VADDR + IOFFSET), so one VDST register serves both chunks.
// Tracked by ASYNCcnt.
__device__ __forceinline__ void async_b_tile_32B(unsigned lds,
                                                 const unsigned short* gsrc) {
  asm volatile("global_load_async_to_lds_b128 %0, %1, off\n\t"
               "global_load_async_to_lds_b128 %0, %1, off offset:32"
               :: "v"(lds), "v"(gsrc) : "memory");
}

// ---------------------------------------------------------------------------
// Generic NT GEMM:  C[M,N] (f32) = A[MxK](bf16, row major) * B[NxK](bf16)^T
// Block = 128 threads (4 waves); workgroup tile 64x64, wave tile 16x64.
// B tile async-copied into double-buffered LDS; fragments via ds_load_b128.
// M,N multiples of 64; K multiple of 32.
// ---------------------------------------------------------------------------
__global__ void __launch_bounds__(128) gemm_nt(
    const unsigned short* __restrict__ A,
    const unsigned short* __restrict__ B,
    float* __restrict__ C,
    const float* __restrict__ bias,
    int M, int N, int K)
{
  __shared__ __align__(16) unsigned short Bs[2][64 * 32];
  const int tid  = threadIdx.x;
  const int lane = tid & 31;
  const int wave = tid >> 5;
  const int n16  = lane & 15;
  const int half = (lane >> 4) & 1;
  const int m0 = blockIdx.y * 64 + wave * 16;
  const int n0 = blockIdx.x * 64;

  v8f c0 = {0,0,0,0,0,0,0,0};
  v8f c1 = c0, c2 = c0, c3 = c0;

  const int rstage = tid >> 1;   // 0..63 : B row within tile
  const int pstage = tid & 1;    // which 16B pair of the 64B row
  const unsigned short* browBase = B + (long)(n0 + rstage) * K + pstage * 8;
  const unsigned short* arowBase = A + (long)(m0 + n16) * K;

  unsigned ldsDst[2];
  ldsDst[0] = (unsigned)(size_t)&Bs[0][rstage * 32 + pstage * 8];
  ldsDst[1] = (unsigned)(size_t)&Bs[1][rstage * 32 + pstage * 8];

  // Preload K-tile 0 into buffer 0.
  async_b_tile_32B(ldsDst[0], browBase);

  for (int k0 = 0; k0 < K; k0 += 32) {
    const int  cur  = (k0 >> 5) & 1;
    const bool more = (k0 + 32) < K;
    if (more) {
      // Kick off the next tile into the other buffer; it lands while the
      // WMMAs below consume the current one.
      async_b_tile_32B(ldsDst[cur ^ 1], browBase + k0 + 32);
      asm volatile("s_wait_asynccnt 2" ::: "memory");  // current tile landed
    } else {
      asm volatile("s_wait_asynccnt 0" ::: "memory");
    }
    __syncthreads();

    Frag a = load_frag_g(arowBase + k0 + half * 8);

    Frag b0, b1, b2, b3;
    {
      const unsigned short* lp0 = &Bs[cur][(0 * 16 + n16) * 32 + half * 8];
      const unsigned short* lp1 = &Bs[cur][(1 * 16 + n16) * 32 + half * 8];
      const unsigned short* lp2 = &Bs[cur][(2 * 16 + n16) * 32 + half * 8];
      const unsigned short* lp3 = &Bs[cur][(3 * 16 + n16) * 32 + half * 8];
      b0.q[0] = *(const uint4*)lp0; b0.q[1] = *(const uint4*)(lp0 + 16);
      b1.q[0] = *(const uint4*)lp1; b1.q[1] = *(const uint4*)(lp1 + 16);
      b2.q[0] = *(const uint4*)lp2; b2.q[1] = *(const uint4*)(lp2 + 16);
      b3.q[0] = *(const uint4*)lp3; b3.q[1] = *(const uint4*)(lp3 + 16);
    }
    c0 = wmma_bf16(a.v, b0.v, c0);
    c1 = wmma_bf16(a.v, b1.v, c1);
    c2 = wmma_bf16(a.v, b2.v, c2);
    c3 = wmma_bf16(a.v, b3.v, c3);
    __syncthreads();   // all waves done reading Bs[cur] before it is refilled
  }

#pragma unroll
  for (int nt = 0; nt < 4; ++nt) {
    v8f cc = nt == 0 ? c0 : nt == 1 ? c1 : nt == 2 ? c2 : c3;
    int n = n0 + nt * 16 + n16;
    float bv = bias ? bias[n] : 0.0f;
#pragma unroll
    for (int g = 0; g < 8; ++g) {
      int m = m0 + g + 8 * half;
      C[(long)m * N + n] = cc[g] + bv;
    }
  }
}

// ---------------------------------------------------------------------------
// Fused flash attention, 1 wave per 16-row query tile, DH = 64.
// Optionally adds the Transformer-XL relative bias: pos[i,j] = scale *
// q[i].rel_w[j - i + 1023], computed with WMMA over the needed 48-row band,
// diagonal-gathered through LDS (matches the reference _shift()).
// K operand is kb (rows x 64); V operand is pre-transposed vtb (64 x len).
// outB != null : write bf16 to (b, t, e=h*64+d) layout (main attention).
// outB == null : write f32 to (bh, t, 64)            (aux attentions).
// ---------------------------------------------------------------------------
__global__ void __launch_bounds__(32) flash_attn(
    const unsigned short* __restrict__ qb,    // (BH, 1024, 64) bf16
    const unsigned short* __restrict__ kb,    // per-bh stride kBh
    const unsigned short* __restrict__ vtb,   // per-bh stride vtBh, row stride vtStride
    const unsigned short* __restrict__ relw,  // (16, 2304, 64) bf16
    float* __restrict__ outF,
    unsigned short* __restrict__ outB,
    int kvLen, long kBh, long vtBh, int vtStride,
    float scale, int hasPos)
{
  const int lane = threadIdx.x;
  const int n16  = lane & 15;
  const int half = (lane >> 4) & 1;
  const int qt = blockIdx.x;
  const int bh = blockIdx.y;
  const int h  = bh & 15;
  const int b  = bh >> 4;
  const int i0 = qt * 16;

  __shared__ __align__(16) float          Tls[16 * 48];
  __shared__ __align__(16) unsigned short Pls[16 * 32];

  const unsigned short* qrow = qb + ((long)bh * 1024 + i0 + n16) * 64;
  Frag qa0 = load_frag_g(qrow + half * 8);
  Frag qa1 = load_frag_g(qrow + 32 + half * 8);

  v8f o0 = {0,0,0,0,0,0,0,0};
  v8f o1 = o0, o2 = o0, o3 = o0;
  float mrow[8], lrow[8];
#pragma unroll
  for (int g = 0; g < 8; ++g) { mrow[g] = -1e30f; lrow[g] = 0.0f; }

  const unsigned short* kbh = kb  + (long)bh * kBh;
  const unsigned short* vbh = vtb + (long)bh * vtBh;

  for (int jc = 0; jc < kvLen; jc += 32) {
    if (jc + 32 < kvLen)
      __builtin_prefetch(kbh + (long)(jc + 32 + n16) * 64, 0, 0);

    v8f s0 = {0,0,0,0,0,0,0,0};
    v8f s1 = s0;
    {
      const unsigned short* kr0 = kbh + (long)(jc + n16) * 64 + half * 8;
      Frag kf0 = load_frag_g(kr0);
      Frag kf1 = load_frag_g(kr0 + 32);
      s0 = wmma_bf16(qa0.v, kf0.v, s0);
      s0 = wmma_bf16(qa1.v, kf1.v, s0);
      const unsigned short* kr1 = kbh + (long)(jc + 16 + n16) * 64 + half * 8;
      Frag kf2 = load_frag_g(kr1);
      Frag kf3 = load_frag_g(kr1 + 32);
      s1 = wmma_bf16(qa0.v, kf2.v, s1);
      s1 = wmma_bf16(qa1.v, kf3.v, s1);
    }
#pragma unroll
    for (int g = 0; g < 8; ++g) { s0[g] *= scale; s1[g] *= scale; }

    if (hasPos) {
      const int rb = 1008 + jc - i0;   // = 1023 + (jc - 15) - i0
#pragma unroll
      for (int tt = 0; tt < 3; ++tt) {
        int rr = rb + tt * 16 + n16;
        int rc = rr < 0 ? 0 : (rr > 2303 ? 2303 : rr);
        const unsigned short* wr = relw + ((long)h * 2304 + rc) * 64 + half * 8;
        Frag wb0 = load_frag_g(wr);
        Frag wb1 = load_frag_g(wr + 32);
        v8f t = {0,0,0,0,0,0,0,0};
        t = wmma_bf16(qa0.v, wb0.v, t);
        t = wmma_bf16(qa1.v, wb1.v, t);
#pragma unroll
        for (int g = 0; g < 8; ++g)
          Tls[(g + 8 * half) * 48 + tt * 16 + n16] = t[g];
      }
      __syncthreads();
#pragma unroll
      for (int g = 0; g < 8; ++g) {
        int ii  = g + 8 * half;
        int c0i = n16 - ii + 15;          // column in the 48-wide band
        int r0  = rb + c0i;
        if (r0 >= 0 && r0 < 2304) s0[g] += scale * Tls[ii * 48 + c0i];
        int c1i = n16 + 31 - ii;
        int r1  = rb + c1i;
        if (r1 >= 0 && r1 < 2304) s1[g] += scale * Tls[ii * 48 + c1i];
      }
      __syncthreads();
    }

    // ---- online softmax over this 32-key slab -----------------------------
    float fac[8];
#pragma unroll
    for (int g = 0; g < 8; ++g) {
      float v = fmaxf(s0[g], s1[g]);
      v = fmaxf(v, __shfl_xor(v, 1, 32));
      v = fmaxf(v, __shfl_xor(v, 2, 32));
      v = fmaxf(v, __shfl_xor(v, 4, 32));
      v = fmaxf(v, __shfl_xor(v, 8, 32));
      float mn = fmaxf(mrow[g], v);
      float f  = __expf(mrow[g] - mn);
      mrow[g] = mn;
      fac[g]  = f;
      s0[g] = __expf(s0[g] - mn);
      s1[g] = __expf(s1[g] - mn);
      float rs = s0[g] + s1[g];
      rs += __shfl_xor(rs, 1, 32);
      rs += __shfl_xor(rs, 2, 32);
      rs += __shfl_xor(rs, 4, 32);
      rs += __shfl_xor(rs, 8, 32);
      lrow[g] = lrow[g] * f + rs;
    }
#pragma unroll
    for (int g = 0; g < 8; ++g) {
      o0[g] *= fac[g]; o1[g] *= fac[g]; o2[g] *= fac[g]; o3[g] *= fac[g];
    }

    // ---- P (C-layout f32) -> bf16 A-layout via LDS ------------------------
#pragma unroll
    for (int g = 0; g < 8; ++g) {
      int ii = g + 8 * half;
      Pls[ii * 32 + n16]      = f2bf(s0[g]);
      Pls[ii * 32 + 16 + n16] = f2bf(s1[g]);
    }
    __syncthreads();
    Frag pa;
    {
      const unsigned short* pp = &Pls[n16 * 32 + half * 8];
      pa.q[0] = *(const uint4*)pp;
      pa.q[1] = *(const uint4*)(pp + 16);
    }
    __syncthreads();

    // ---- O += P @ V (B fragments contiguous from V^T) ---------------------
    {
      const unsigned short* vr0 = vbh + (long)(0 * 16 + n16) * vtStride + jc + half * 8;
      const unsigned short* vr1 = vbh + (long)(1 * 16 + n16) * vtStride + jc + half * 8;
      const unsigned short* vr2 = vbh + (long)(2 * 16 + n16) * vtStride + jc + half * 8;
      const unsigned short* vr3 = vbh + (long)(3 * 16 + n16) * vtStride + jc + half * 8;
      Frag vb0 = load_frag_g(vr0);
      Frag vb1 = load_frag_g(vr1);
      Frag vb2 = load_frag_g(vr2);
      Frag vb3 = load_frag_g(vr3);
      o0 = wmma_bf16(pa.v, vb0.v, o0);
      o1 = wmma_bf16(pa.v, vb1.v, o1);
      o2 = wmma_bf16(pa.v, vb2.v, o2);
      o3 = wmma_bf16(pa.v, vb3.v, o3);
    }
  }

#pragma unroll
  for (int g = 0; g < 8; ++g) lrow[g] = 1.0f / lrow[g];

#pragma unroll
  for (int nt = 0; nt < 4; ++nt) {
    v8f oc = nt == 0 ? o0 : nt == 1 ? o1 : nt == 2 ? o2 : o3;
#pragma unroll
    for (int g = 0; g < 8; ++g) {
      int ii = g + 8 * half;
      int dd = nt * 16 + n16;
      float v = oc[g] * lrow[g];
      if (outB)
        outB[((long)b * 1024 + i0 + ii) * 1024 + h * 64 + dd] = f2bf(v);
      else
        outF[((long)bh * 1024 + i0 + ii) * 64 + dd] = v;
    }
  }
}

// ---------------------------------------------------------------------------
// Elementwise / layout kernels
// ---------------------------------------------------------------------------
__global__ void cast_f32_bf16_k(const float* __restrict__ in,
                                unsigned short* __restrict__ out, long n) {
  long i = (long)blockIdx.x * blockDim.x + threadIdx.x;
  if (i < n) out[i] = f2bf(in[i]);
}

__global__ void copy_f32_k(const float* __restrict__ in,
                           float* __restrict__ out, long n) {
  long i = (long)blockIdx.x * blockDim.x + threadIdx.x;
  if (i < n) out[i] = in[i];
}

// kv_input = concat([cmem(256), mem(1024), x(1024)], axis=1) -> bf16 (4,2304,1024)
__global__ void concat_kvin_k(const float* __restrict__ cm,
                              const float* __restrict__ mm,
                              const float* __restrict__ xx,
                              unsigned short* __restrict__ out) {
  long i = (long)blockIdx.x * blockDim.x + threadIdx.x;
  long n = 4L * 2304 * 1024;
  if (i >= n) return;
  int  e  = (int)(i & 1023);
  long je = i >> 10;
  int  j  = (int)(je % 2304);
  int  b  = (int)(je / 2304);
  float v;
  if (j < 256)       v = cm[((long)b * 256  + j)        * 1024 + e];
  else if (j < 1280) v = mm[((long)b * 1024 + (j - 256))  * 1024 + e];
  else               v = xx[((long)b * 1024 + (j - 1280)) * 1024 + e];
  out[i] = f2bf(v);
}

// (K,N) f32 -> (N,K) bf16
__global__ void transpose_cast_k(const float* __restrict__ in,
                                 unsigned short* __restrict__ out,
                                 int K, int N) {
  long i = (long)blockIdx.x * blockDim.x + threadIdx.x;
  long n = (long)K * N;
  if (i >= n) return;
  int  k = (int)(i % K);
  long r = i / K;               // output row (0..N-1)
  out[i] = f2bf(in[(long)k * N + r]);
}

// conv_w (o=1024, i=1024, r=4) -> out[o][r*1024+i]
__global__ void convw_cast_k(const float* __restrict__ in,
                             unsigned short* __restrict__ out) {
  long i = (long)blockIdx.x * blockDim.x + threadIdx.x;
  long n = 1024L * 4096;
  if (i >= n) return;
  int o  = (int)(i >> 12);
  int k  = (int)(i & 4095);
  int r  = k >> 10;
  int ii = k & 1023;
  out[i] = f2bf(in[((long)o * 1024 + ii) * 4 + r]);
}

// rel_w (2304,16,64) -> (16,2304,64) bf16
__global__ void relw_cast_k(const float* __restrict__ in,
                            unsigned short* __restrict__ out) {
  long i = (long)blockIdx.x * blockDim.x + threadIdx.x;
  long n = 16L * 2304 * 64;
  if (i >= n) return;
  int  d = (int)(i & 63);
  long t = i >> 6;
  int  r = (int)(t % 2304);
  int  h = (int)(t / 2304);
  out[i] = f2bf(in[((long)r * 16 + h) * 64 + d]);
}

// q (b,1024,16*64) f32 -> (b,16,1024,64) bf16
__global__ void q_split_k(const float* __restrict__ in,
                          unsigned short* __restrict__ out) {
  long i = (long)blockIdx.x * blockDim.x + threadIdx.x;
  long n = 4L * 16 * 1024 * 64;
  if (i >= n) return;
  int  d = (int)(i & 63);
  long t = i >> 6;
  int  row = (int)(t % 1024); t /= 1024;
  int  h   = (int)(t % 16);
  int  b   = (int)(t / 16);
  out[i] = f2bf(in[((long)b * 1024 + row) * 1024 + h * 64 + d]);
}

// kv (b,L,2048) f32 -> k (b,16,L,64) bf16 and v^T (b,16,64,L) bf16
__global__ void kv_split_k(const float* __restrict__ kv,
                           unsigned short* __restrict__ kout,
                           unsigned short* __restrict__ vtout, int L) {
  long i = (long)blockIdx.x * blockDim.x + threadIdx.x;
  long n = 4L * 16 * L * 64;
  if (i >= n) return;
  int  d = (int)(i & 63);
  long t = i >> 6;
  int  j = (int)(t % L); t /= L;
  int  h = (int)(t % 16);
  int  b = (int)(t / 16);
  const float* src = kv + ((long)b * L + j) * 2048 + h * 64 + d;
  kout[i] = f2bf(src[0]);
  vtout[(((long)(b * 16 + h) * 64 + d) * L) + j] = f2bf(src[1024]);
}

// ---------------------------------------------------------------------------
// aux_loss = mean((a_old - a_cmp)^2) : two-stage reduction
// ---------------------------------------------------------------------------
__global__ void __launch_bounds__(256) loss_partial_k(
    const float* __restrict__ a, const float* __restrict__ c,
    float* __restrict__ part, long n) {
  __shared__ float sm[256];
  float s = 0.0f;
  for (long i = (long)blockIdx.x * 256 + threadIdx.x; i < n;
       i += (long)gridDim.x * 256) {
    float d = a[i] - c[i];
    s += d * d;
  }
  sm[threadIdx.x] = s;
  __syncthreads();
  for (int st = 128; st > 0; st >>= 1) {
    if ((int)threadIdx.x < st) sm[threadIdx.x] += sm[threadIdx.x + st];
    __syncthreads();
  }
  if (threadIdx.x == 0) part[blockIdx.x] = sm[0];
}

__global__ void __launch_bounds__(256) loss_final_k(
    const float* __restrict__ part, float* __restrict__ out,
    int np, float invN) {
  __shared__ float sm[256];
  float s = 0.0f;
  for (int i = threadIdx.x; i < np; i += 256) s += part[i];
  sm[threadIdx.x] = s;
  __syncthreads();
  for (int st = 128; st > 0; st >>= 1) {
    if ((int)threadIdx.x < st) sm[threadIdx.x] += sm[threadIdx.x + st];
    __syncthreads();
  }
  if (threadIdx.x == 0) out[0] = sm[0] * invN;
}

// ---------------------------------------------------------------------------
// Host orchestration
// ---------------------------------------------------------------------------
static inline dim3 grid1d(long n) { return dim3((unsigned)((n + 255) / 256)); }

extern "C" void kernel_launch(void* const* d_in, const int* in_sizes, int n_in,
                              void* d_out, int out_size, void* d_ws, size_t ws_size,
                              hipStream_t stream) {
  (void)in_sizes; (void)n_in; (void)out_size; (void)ws_size;

  const float* x      = (const float*)d_in[0];
  const float* mem    = (const float*)d_in[1];
  const float* cmem   = (const float*)d_in[2];
  const float* Wq     = (const float*)d_in[3];
  const float* Wkv    = (const float*)d_in[4];
  const float* Wo     = (const float*)d_in[5];
  const float* bo     = (const float*)d_in[6];
  const float* rel_w  = (const float*)d_in[7];
  const float* conv_w = (const float*)d_in[8];
  const float* conv_b = (const float*)d_in[9];
  float* out = (float*)d_out;

  const long N_xe   = 4L * 1024 * 1024;       // b*t*e
  const long N_kvin = 4L * 2304 * 1024;
  const long N_cme  = 4L * 256 * 1024;
  const long OUT_OFF  = 0;
  const long MEM_OFF  = N_xe;                 // new_mem
  const long CMEM_OFF = MEM_OFF + N_xe;       // new_cmem (conv GEMM writes here)
  const long LOSS_OFF = CMEM_OFF + N_cme;     // scalar aux_loss

  // ---- workspace carve-up (256B aligned) ----
  char*  w   = (char*)d_ws;
  size_t off = 0;
  auto alloc = [&](size_t bytes) {
    void* p = w + off;
    off = (off + bytes + 255) & ~(size_t)255;
    return p;
  };
  unsigned short* x_bf    = (unsigned short*)alloc(N_xe * 2);
  unsigned short* mem_bf  = (unsigned short*)alloc(N_xe * 2);
  unsigned short* kvin_bf = (unsigned short*)alloc(N_kvin * 2);
  unsigned short* wq_t    = (unsigned short*)alloc(1024L * 1024 * 2);
  unsigned short* wkv_t   = (unsigned short*)alloc(2048L * 1024 * 2);
  unsigned short* wo_t    = (unsigned short*)alloc(1024L * 1024 * 2);
  unsigned short* convw_t = (unsigned short*)alloc(1024L * 4096 * 2);
  unsigned short* relw_r  = (unsigned short*)alloc(16L * 2304 * 64 * 2);
  float*          q_f     = (float*)alloc(N_xe * 4);
  unsigned short* q_bf    = (unsigned short*)alloc(N_xe * 2);
  float*          kv_f    = (float*)alloc(4L * 2304 * 2048 * 4);
  unsigned short* k_bf    = (unsigned short*)alloc(N_kvin * 2);
  unsigned short* vt_bf   = (unsigned short*)alloc(N_kvin * 2);
  unsigned short* comp_bf = (unsigned short*)alloc(N_cme * 2);
  float*          ckv_f   = (float*)alloc(4L * 256 * 2048 * 4);
  unsigned short* ck_bf   = (unsigned short*)alloc(4L * 16 * 256 * 64 * 2);
  unsigned short* cvt_bf  = (unsigned short*)alloc(4L * 16 * 256 * 64 * 2);
  unsigned short* attn_bf = (unsigned short*)alloc(N_xe * 2);
  float*          aold_f  = (float*)alloc(N_xe * 4);
  float*          acmp_f  = (float*)alloc(N_xe * 4);
  float*          part_f  = (float*)alloc(1024 * 4);

  // ---- operand preparation (f32 -> bf16, layout changes) ----
  cast_f32_bf16_k<<<grid1d(N_xe), 256, 0, stream>>>(x,   x_bf,   N_xe);
  cast_f32_bf16_k<<<grid1d(N_xe), 256, 0, stream>>>(mem, mem_bf, N_xe);
  concat_kvin_k<<<grid1d(N_kvin), 256, 0, stream>>>(cmem, mem, x, kvin_bf);
  transpose_cast_k<<<grid1d(1024L * 1024), 256, 0, stream>>>(Wq,  wq_t,  1024, 1024);
  transpose_cast_k<<<grid1d(2048L * 1024), 256, 0, stream>>>(Wkv, wkv_t, 1024, 2048);
  transpose_cast_k<<<grid1d(1024L * 1024), 256, 0, stream>>>(Wo,  wo_t,  1024, 1024);
  convw_cast_k<<<grid1d(1024L * 4096), 256, 0, stream>>>(conv_w, convw_t);
  relw_cast_k<<<grid1d(16L * 2304 * 64), 256, 0, stream>>>(rel_w, relw_r);

  // ---- projections (WMMA GEMMs) ----
  gemm_nt<<<dim3(1024 / 64, 4096 / 64), 128, 0, stream>>>(
      x_bf, wq_t, q_f, nullptr, 4096, 1024, 1024);             // Q = x @ Wq
  q_split_k<<<grid1d(N_xe), 256, 0, stream>>>(q_f, q_bf);

  gemm_nt<<<dim3(2048 / 64, 9216 / 64), 128, 0, stream>>>(
      kvin_bf, wkv_t, kv_f, nullptr, 9216, 2048, 1024);        // KV = kv_in @ Wkv
  kv_split_k<<<grid1d(4L * 16 * 2304 * 64), 256, 0, stream>>>(kv_f, k_bf, vt_bf, 2304);

  // ---- compression conv as GEMM: new_cmem straight into d_out ----
  gemm_nt<<<dim3(1024 / 64, 1024 / 64), 128, 0, stream>>>(
      mem_bf, convw_t, out + CMEM_OFF, conv_b, 1024, 1024, 4096);
  cast_f32_bf16_k<<<grid1d(N_cme), 256, 0, stream>>>(out + CMEM_OFF, comp_bf, N_cme);

  gemm_nt<<<dim3(2048 / 64, 1024 / 64), 128, 0, stream>>>(
      comp_bf, wkv_t, ckv_f, nullptr, 1024, 2048, 1024);       // ckv = comp @ Wkv
  kv_split_k<<<grid1d(4L * 16 * 256 * 64), 256, 0, stream>>>(ckv_f, ck_bf, cvt_bf, 256);

  // ---- main attention (scaled, rel-pos bias, full 2304 keys) ----
  flash_attn<<<dim3(64, 64), 32, 0, stream>>>(
      q_bf, k_bf, vt_bf, relw_r, nullptr, attn_bf,
      2304, 2304L * 64, 64L * 2304, 2304, 0.125f, 1);

  // ---- output projection: out = attn @ Wo + bo ----
  gemm_nt<<<dim3(1024 / 64, 4096 / 64), 128, 0, stream>>>(
      attn_bf, wo_t, out + OUT_OFF, bo, 4096, 1024, 1024);

  // ---- auxiliary attentions (unscaled, no pos) ----
  flash_attn<<<dim3(64, 64), 32, 0, stream>>>(           // a_old: k/v rows 256..1279
      q_bf, k_bf + 256L * 64, vt_bf + 256, nullptr, aold_f, nullptr,
      1024, 2304L * 64, 64L * 2304, 2304, 1.0f, 0);
  flash_attn<<<dim3(64, 64), 32, 0, stream>>>(           // a_cmp: compressed k/v
      q_bf, ck_bf, cvt_bf, nullptr, acmp_f, nullptr,
      256, 256L * 64, 64L * 256, 256, 1.0f, 0);

  // ---- new_mem = x ----
  copy_f32_k<<<grid1d(N_xe), 256, 0, stream>>>(x, out + MEM_OFF, N_xe);

  // ---- aux_loss = mean((a_old - a_cmp)^2) ----
  loss_partial_k<<<1024, 256, 0, stream>>>(aold_f, acmp_f, part_f, N_xe);
  loss_final_k<<<1, 256, 0, stream>>>(part_f, out + LOSS_OFF, 1024,
                                      1.0f / (float)N_xe);
}
